// ResBlock_2259152798072
// MI455X (gfx1250) — compile-verified, matrix operand order checked by reading, not run on hardware
//
#include <hip/hip_runtime.h>
#include <stdint.h>

typedef unsigned short u16;
typedef __attribute__((ext_vector_type(16))) __bf16 v16bf;
typedef __attribute__((ext_vector_type(8)))  float  v8f;
typedef __attribute__((ext_vector_type(4)))  unsigned int u32x4;
typedef __attribute__((ext_vector_type(8)))  unsigned int u32x8;

#define BB 32
#define CC 256
#define LL 1024
#define HH 512
#define NN 64
#define HN_ (HH*NN)
#define MM 32768   // B*L

// ---------- helpers ----------
__device__ __forceinline__ u16 f2bf(float f) {
  union { float f; uint32_t u; } x; x.f = f;
  return (u16)((x.u + 0x7FFFu + ((x.u >> 16) & 1u)) >> 16);
}
__device__ __forceinline__ float sigmf(float x) { return 1.f / (1.f + expf(-x)); }
__device__ __forceinline__ float geluf(float x) {
  float x3 = x * x * x;
  return 0.5f * x * (1.f + tanhf(0.7978845608028654f * (x + 0.044715f * x3)));
}
__device__ __forceinline__ void store_val(float* p, float v) { *p = v; }
__device__ __forceinline__ void store_val(u16* p, float v) { *p = f2bf(v); }

// LDS byte offset of a __shared__ object (generic -> addrspace(3) -> int)
__device__ __forceinline__ unsigned lds_off(const void* p) {
  return (unsigned)(unsigned long long)(__attribute__((address_space(3))) const void*)p;
}

// ---------- Tensor Data Mover: 2D tile (rows x 32 bf16) global->LDS ----------
// LDS padding: 16 data DWORDs (64B row) + 4 pad DWORDs -> row stride 40 u16 (80B),
// matching the bank-conflict-free fragment layout below.
__device__ __forceinline__ void tdm_load_tile(unsigned lds_byte, const u16* gptr,
                                              unsigned K, unsigned rows_total,
                                              unsigned tile_rows) {
  unsigned long long ga = (unsigned long long)gptr;
  u32x4 g0;
  g0[0] = 1u;                                            // count=1 valid descriptor
  g0[1] = lds_byte;                                      // lds_addr
  g0[2] = (unsigned)ga;                                  // global_addr[31:0]
  g0[3] = (unsigned)((ga >> 32) & 0x1FFFFFFu) | (2u << 30);  // addr[56:32] | type=2
  u32x8 g1;
  // data_size=1 (2B) | pad_enable | pad_interval=3 (16 DW) | pad_amount=3 (4 DW)
  g1[0] = (1u << 16) | (1u << 20) | (3u << 22) | (3u << 25);
  g1[1] = (K & 0xFFFFu) << 16;                           // barrier_addr=0 | tensor_dim0 lo
  g1[2] = (K >> 16) | ((rows_total & 0xFFFFu) << 16);    // tensor_dim0 hi | tensor_dim1 lo
  g1[3] = (rows_total >> 16) | (32u << 16);              // tensor_dim1 hi | tile_dim0=32
  g1[4] = tile_rows;                                     // tile_dim1 | tile_dim2=0
  g1[5] = K;                                             // tensor_dim0_stride[31:0] (elems)
  g1[6] = 0u;                                            // stride hi | dim1_stride lo
  g1[7] = 0u;
  asm volatile("tensor_load_to_lds %0, %1" :: "s"(g0), "s"(g1) : "memory");
}

// ---------- S4 discretization: w = exp(dt*A), Cd = C*(exp(dt*A)-1)/A ----------
__global__ void discretize(const float* log_dt, const float* log_A_real, const float* A_imag,
                           const float* C_re, const float* C_im, float* par) {
  int idx = blockIdx.x * blockDim.x + threadIdx.x;
  if (idx >= HN_) return;
  int h = idx / NN;
  float dt = expf(log_dt[h]);
  float ar = -expf(log_A_real[idx]);
  float ai = A_imag[idx];
  float dr = ar * dt, di = ai * dt;
  float e = expf(dr);
  float wr = e * cosf(di), wi = e * sinf(di);
  float nr = wr - 1.f, ni = wi;
  float den = ar * ar + ai * ai;
  float fr = (nr * ar + ni * ai) / den;
  float fi = (ni * ar - nr * ai) / den;
  par[idx] = wr; par[HN_ + idx] = wi;
  for (int d = 0; d < 2; ++d) {
    float cr = C_re[d * HN_ + idx], ci = C_im[d * HN_ + idx];
    par[(2 + 2 * d) * HN_ + idx] = cr * fr - ci * fi;
    par[(3 + 2 * d) * HN_ + idx] = cr * fi + ci * fr;
  }
}

// ---------- embedding GEMV: emb[b,c] = diff[b,:].W[c,:] + b[c] ----------
__global__ void emb_kernel(const float* demb, const float* W, const float* bias, float* out) {
  int idx = blockIdx.x * blockDim.x + threadIdx.x;  // 32*256
  int bi = idx >> 8, c = idx & 255;
  float s = bias[c];
  const float* dp = demb + (size_t)bi * 512;
  const float* wp = W + (size_t)c * 512;
  for (int e = 0; e < 512; ++e) s += dp[e] * wp[e];
  out[idx] = s;
}

// ---------- im2col for conv3 (pad 1), with emb broadcast add, bf16 ----------
__global__ void pack_im2col(const float* x_t, const float* emb, u16* A) {
  size_t idx = (size_t)blockIdx.x * blockDim.x + threadIdx.x;  // MM*768
  int m = (int)(idx / 768);
  int kk = (int)(idx % 768);
  int k3 = kk >> 8, c = kk & 255;
  int bi = m >> 10, l = m & 1023;
  int ll = l + k3 - 1;
  float v = 0.f;
  if (ll >= 0 && ll < LL) v = x_t[((size_t)bi * CC + c) * LL + ll] + emb[bi * CC + c];
  A[idx] = f2bf(v);
}

// ---------- conv3 weight pack: Bw[cout, k3*256+cin] = W[cout,cin,k3] ----------
__global__ void pack_w_mid(const float* W, u16* Bm) {
  int idx = blockIdx.x * blockDim.x + threadIdx.x;  // 512*768
  int n = idx / 768, kk = idx % 768;
  int k3 = kk >> 8, c = kk & 255;
  Bm[idx] = f2bf(W[(size_t)n * 768 + c * 3 + k3]);
}

__global__ void copy_bf16(const float* in, u16* out, int n) {
  int idx = blockIdx.x * blockDim.x + threadIdx.x;
  if (idx < n) out[idx] = f2bf(in[idx]);
}

__global__ void zero_kernel(float* p, int n) {
  int i = blockIdx.x * blockDim.x + threadIdx.x;
  if (i < n) p[i] = 0.f;
}

// ---------- WMMA GEMM: C[M,N] = A[M,K](bf16) * Bw[N,K]^T(bf16) + bias ----------
__device__ __forceinline__ v16bf ld_frag(const u16* t, int row, int lane) {
  // ISA 7.12.2: 16-bit A/B fragment; lanes 0-15 K={0..7,16..23}, lanes 16-31 K={8..15,24..31}
  int r = row + (lane & 15);
  int hi = (lane >> 4) * 8;
  const u16* p = t + r * 40;
  uint4 q0 = *(const uint4*)(p + hi);
  uint4 q1 = *(const uint4*)(p + 16 + hi);
  union { uint32_t w[8]; v16bf v; } u;
  u.w[0] = q0.x; u.w[1] = q0.y; u.w[2] = q0.z; u.w[3] = q0.w;
  u.w[4] = q1.x; u.w[5] = q1.y; u.w[6] = q1.z; u.w[7] = q1.w;
  return u.v;
}

__global__ __launch_bounds__(256) void gemm_bf16(const u16* __restrict__ A, const u16* __restrict__ Bw,
                                                 float* __restrict__ Cm, const float* __restrict__ bias,
                                                 int M, int N, int K) {
  __shared__ u16 As[2][64][40];    // 64 rows x 32 k, stride 40 (TDM-padded, conflict-free)
  __shared__ u16 Bs[2][128][40];   // 128 n-rows x 32 k
  int tid = threadIdx.x;
  int lane = tid & 31, w = tid >> 5;
  int wm = (w >> 2) * 32, wn = (w & 3) * 32;
  size_t m0 = (size_t)blockIdx.y * 64;
  size_t n0 = (size_t)blockIdx.x * 128;
  v8f zero8 = {0.f, 0.f, 0.f, 0.f, 0.f, 0.f, 0.f, 0.f};
  v8f acc[2][2]; acc[0][0] = zero8; acc[0][1] = zero8; acc[1][0] = zero8; acc[1][1] = zero8;
  int nk = K >> 5;
  bool issuer = (tid < 32);        // wave 0 drives the Tensor Data Mover
  unsigned ldsA0 = lds_off(&As[0][0][0]), ldsA1 = lds_off(&As[1][0][0]);
  unsigned ldsB0 = lds_off(&Bs[0][0][0]), ldsB1 = lds_off(&Bs[1][0][0]);
  if (issuer) {
    tdm_load_tile(ldsA0, A + m0 * K, (unsigned)K, (unsigned)M, 64u);
    tdm_load_tile(ldsB0, Bw + n0 * K, (unsigned)K, (unsigned)N, 128u);
  }
  for (int kt = 0; kt < nk; ++kt) {
    int buf = kt & 1;
    if (issuer) {
      if (kt + 1 < nk) {
        int k = (kt + 1) << 5;
        tdm_load_tile(buf ? ldsA0 : ldsA1, A + m0 * K + k, (unsigned)K, (unsigned)M, 64u);
        tdm_load_tile(buf ? ldsB0 : ldsB1, Bw + n0 * K + k, (unsigned)K, (unsigned)N, 128u);
        // TDM completes in order per wave: <=2 outstanding => stage kt has landed
        __builtin_amdgcn_s_wait_tensorcnt(2);
      } else {
        __builtin_amdgcn_s_wait_tensorcnt(0);
      }
      if (kt + 2 < nk)
        __builtin_prefetch(A + m0 * K + ((kt + 2) << 5) + lane * 64, 0, 1);
    }
    __syncthreads();               // publish DMA'd tiles to all 8 waves
    v16bf a0 = ld_frag(&As[buf][0][0], wm, lane);
    v16bf a1 = ld_frag(&As[buf][0][0], wm + 16, lane);
    v16bf b0 = ld_frag(&Bs[buf][0][0], wn, lane);
    v16bf b1 = ld_frag(&Bs[buf][0][0], wn + 16, lane);
    acc[0][0] = __builtin_amdgcn_wmma_f32_16x16x32_bf16(false, a0, false, b0, (short)0, acc[0][0], false, false);
    acc[0][1] = __builtin_amdgcn_wmma_f32_16x16x32_bf16(false, a0, false, b1, (short)0, acc[0][1], false, false);
    acc[1][0] = __builtin_amdgcn_wmma_f32_16x16x32_bf16(false, a1, false, b0, (short)0, acc[1][0], false, false);
    acc[1][1] = __builtin_amdgcn_wmma_f32_16x16x32_bf16(false, a1, false, b1, (short)0, acc[1][1], false, false);
    __syncthreads();               // all waves done with buf before TDM overwrites it
  }
  // C layout: lanes 0-15 -> M=r, N=lane ; lanes 16-31 -> M=8+r, N=lane-16
  int r16 = lane & 15, hi = lane >> 4;
  for (int i = 0; i < 2; ++i)
    for (int j = 0; j < 2; ++j) {
      size_t mmb = m0 + wm + i * 16 + hi * 8;
      size_t nn = n0 + wn + j * 16 + r16;
      float bv = bias ? bias[nn] : 0.f;
#pragma unroll
      for (int r = 0; r < 8; ++r)
        Cm[(mmb + r) * N + nn] = acc[i][j][r] + bv;
    }
}

// ---------- BatchNorm (training-mode batch stats) ----------
__global__ void bn_stats(const float* Cm, float* stats, int N) {
  int n = blockIdx.x * 256 + threadIdx.x;
  int r0 = blockIdx.y * 128;
  float s = 0.f, s2 = 0.f;
  for (int r = 0; r < 128; ++r) {
    float v = Cm[(size_t)(r0 + r) * N + n];
    s += v; s2 += v * v;
  }
  atomicAdd(&stats[n], s);
  atomicAdd(&stats[1024 + n], s2);
}

__global__ void bn_apply_row(const float* Cm, const float* stats, const float* g, const float* be,
                             float* X, int N, float invM, int addmode) {
  size_t idx = (size_t)blockIdx.x * blockDim.x + threadIdx.x;  // M*N
  int n = (int)(idx % N);
  float mu = stats[n] * invM;
  float var = stats[1024 + n] * invM - mu * mu;
  float bn = (Cm[idx] - mu) * rsqrtf(var + 1e-5f) * g[n] + be[n];
  X[idx] = addmode ? (X[idx] + bn) : bn;
}

__global__ void bn_apply_tr(const float* Cm, const float* stats, const float* g, const float* be,
                            const float* xres, float* out, int N, float invM, float scale) {
  size_t idx = (size_t)blockIdx.x * blockDim.x + threadIdx.x;  // B*N*L
  int l = (int)(idx % LL);
  int c = (int)((idx / LL) % N);
  int bi = (int)(idx / ((size_t)N * LL));
  size_t m = (size_t)bi * LL + l;
  float mu = stats[c] * invM;
  float var = stats[1024 + c] * invM - mu * mu;
  float bn = (Cm[m * N + c] - mu) * rsqrtf(var + 1e-5f) * g[c] + be[c];
  float r = bn;
  if (xres) r = (xres[idx] + bn) * scale;
  out[idx] = r;
}

// ---------- LDS-tiled per-batch transpose (R,C)->(C,R), optional gelu + bf16 ----------
template <typename OutT, bool GELU>
__global__ void transpose_kernel(const float* in, OutT* out, int R, int Cc) {
  __shared__ float tile[32][33];
  int b = blockIdx.z;
  int c0 = blockIdx.x * 32, r0 = blockIdx.y * 32;
  const float* ip = in + (size_t)b * R * Cc;
  for (int i = threadIdx.y; i < 32; i += 8)
    tile[i][threadIdx.x] = ip[(size_t)(r0 + i) * Cc + c0 + threadIdx.x];
  __syncthreads();
  OutT* op = out + (size_t)b * R * Cc;
  for (int i = threadIdx.y; i < 32; i += 8) {
    float v = tile[threadIdx.x][i];
    if (GELU) v = geluf(v);
    store_val(&op[(size_t)(c0 + i) * R + r0 + threadIdx.x], v);
  }
}

// ---------- S4 forward scan: one wave per (b,h); 64 complex states, 2/lane ----------
__global__ __launch_bounds__(256) void scan_fwd(const float* U, const float* par, const float* Dv, float* P) {
  int wid = ((blockIdx.x * blockDim.x + threadIdx.x) >> 5);
  int lane = threadIdx.x & 31;
  int b = wid >> 9, h = wid & (HH - 1);
  int i0 = h * NN + lane, i1 = i0 + 32;
  float wr0 = par[i0], wi0 = par[HN_ + i0], wr1 = par[i1], wi1 = par[HN_ + i1];
  float c0r = par[2 * HN_ + i0], c0i = par[3 * HN_ + i0];
  float c1r = par[2 * HN_ + i1], c1i = par[3 * HN_ + i1];
  float dd = Dv[h];
  float x0r = 0.f, x0i = 0.f, x1r = 0.f, x1i = 0.f;
  size_t base = ((size_t)b * HH + h) * LL;
  for (int l0 = 0; l0 < LL; l0 += 32) {
    float uv = U[base + l0 + lane];
    float ysave = 0.f;
    for (int j = 0; j < 32; ++j) {
      float u = __shfl(uv, j, 32);
      float nr = wr0 * x0r - wi0 * x0i + u; x0i = wr0 * x0i + wi0 * x0r; x0r = nr;
      nr = wr1 * x1r - wi1 * x1i + u; x1i = wr1 * x1i + wi1 * x1r; x1r = nr;
      float p = c0r * x0r - c0i * x0i + c1r * x1r - c1i * x1i;
      p += __shfl_xor(p, 1, 32); p += __shfl_xor(p, 2, 32); p += __shfl_xor(p, 4, 32);
      p += __shfl_xor(p, 8, 32); p += __shfl_xor(p, 16, 32);
      float y = 2.f * p + u * dd;
      if (lane == j) ysave = y;
    }
    P[base + l0 + lane] = ysave;
  }
}

// ---------- S4 backward (anti-causal) scan: P += y_bwd ----------
__global__ __launch_bounds__(256) void scan_bwd(const float* U, const float* par, float* P) {
  int wid = ((blockIdx.x * blockDim.x + threadIdx.x) >> 5);
  int lane = threadIdx.x & 31;
  int b = wid >> 9, h = wid & (HH - 1);
  int i0 = h * NN + lane, i1 = i0 + 32;
  float wr0 = par[i0], wi0 = par[HN_ + i0], wr1 = par[i1], wi1 = par[HN_ + i1];
  float c0r = par[4 * HN_ + i0], c0i = par[5 * HN_ + i0];
  float c1r = par[4 * HN_ + i1], c1i = par[5 * HN_ + i1];
  float x0r = 0.f, x0i = 0.f, x1r = 0.f, x1i = 0.f;
  size_t base = ((size_t)b * HH + h) * LL;
  for (int l0 = LL - 32; l0 >= 0; l0 -= 32) {
    float uv = U[base + l0 + lane];
    float pv = P[base + l0 + lane];
    for (int j = 31; j >= 0; --j) {
      // output excludes u[l] (state holds u[t>l]), then absorb u[l]
      float p = c0r * x0r - c0i * x0i + c1r * x1r - c1i * x1i;
      p += __shfl_xor(p, 1, 32); p += __shfl_xor(p, 2, 32); p += __shfl_xor(p, 4, 32);
      p += __shfl_xor(p, 8, 32); p += __shfl_xor(p, 16, 32);
      if (lane == j) pv += 2.f * p;
      float u = __shfl(uv, j, 32);
      float nr = wr0 * x0r - wi0 * x0i + u; x0i = wr0 * x0i + wi0 * x0r; x0r = nr;
      nr = wr1 * x1r - wi1 * x1i + u; x1i = wr1 * x1i + wi1 * x1r; x1r = nr;
    }
    P[base + l0 + lane] = pv;
  }
}

// ---------- GLU + residual + LayerNorm (one block per (b,l) row) ----------
__global__ __launch_bounds__(256) void glu_res_ln(const float* Cm, const float* lng, const float* lnb, float* X) {
  __shared__ float rs[256], rs2[256];
  int m = blockIdx.x, tid = threadIdx.x;
  const float* row = Cm + (size_t)m * 1024;
  float* xr = X + (size_t)m * 512;
  float t0, t1, s = 0.f, s2 = 0.f;
  {
    float v = row[tid], g = row[512 + tid];
    t0 = v * sigmf(g) + xr[tid];
    s += t0; s2 += t0 * t0;
  }
  {
    float v = row[tid + 256], g = row[768 + tid];
    t1 = v * sigmf(g) + xr[tid + 256];
    s += t1; s2 += t1 * t1;
  }
  rs[tid] = s; rs2[tid] = s2; __syncthreads();
  for (int o = 128; o > 0; o >>= 1) {
    if (tid < o) { rs[tid] += rs[tid + o]; rs2[tid] += rs2[tid + o]; }
    __syncthreads();
  }
  float mu = rs[0] * (1.f / 512.f);
  float var = rs2[0] * (1.f / 512.f) - mu * mu;
  float rstd = rsqrtf(var + 1e-5f);
  xr[tid] = (t0 - mu) * rstd * lng[tid] + lnb[tid];
  xr[tid + 256] = (t1 - mu) * rstd * lng[tid + 256] + lnb[tid + 256];
}

// ---------- sigmoid(gate)*tanh(filt), pack bf16 for res/skip GEMMs ----------
__global__ void glu_gate(const float* X, u16* A) {
  size_t idx = (size_t)blockIdx.x * blockDim.x + threadIdx.x;  // MM*256
  int m = (int)(idx >> 8), c = (int)(idx & 255);
  float g = X[(size_t)m * 512 + c];
  float f = X[(size_t)m * 512 + 256 + c];
  A[idx] = f2bf(sigmf(g) * tanhf(f));
}

// =====================================================================
extern "C" void kernel_launch(void* const* d_in, const int* in_sizes, int n_in,
                              void* d_out, int out_size, void* d_ws, size_t ws_size,
                              hipStream_t stream) {
  (void)in_sizes; (void)n_in; (void)out_size; (void)ws_size;
  const float* x_t    = (const float*)d_in[0];
  const float* x_con  = (const float*)d_in[1];
  const float* demb   = (const float*)d_in[2];
  const float* embW   = (const float*)d_in[3];
  const float* embb   = (const float*)d_in[4];
  const float* condW  = (const float*)d_in[5];
  const float* condb  = (const float*)d_in[6];
  const float* condg  = (const float*)d_in[7];
  const float* condbe = (const float*)d_in[8];
  const float* midW   = (const float*)d_in[9];
  const float* midb   = (const float*)d_in[10];
  const float* midg   = (const float*)d_in[11];
  const float* midbe  = (const float*)d_in[12];
  const float* s4a[10]; for (int i = 0; i < 10; ++i) s4a[i] = (const float*)d_in[13 + i];
  const float* s4b[10]; for (int i = 0; i < 10; ++i) s4b[i] = (const float*)d_in[23 + i];
  const float* resW   = (const float*)d_in[33];
  const float* resb   = (const float*)d_in[34];
  const float* resg   = (const float*)d_in[35];
  const float* resbe  = (const float*)d_in[36];
  const float* skipW  = (const float*)d_in[37];
  const float* skipb  = (const float*)d_in[38];
  const float* skipg  = (const float*)d_in[39];
  const float* skipbe = (const float*)d_in[40];
  float* dout = (float*)d_out;

  // workspace arena (~390 MB peak)
  char* ws = (char*)d_ws;
  size_t off = 0;
  auto take = [&](size_t bytes) -> char* {
    char* p = ws + off; off += (bytes + 255) & ~(size_t)255; return p;
  };
  float* PARa  = (float*)take((size_t)6 * HN_ * 4);
  float* PARb  = (float*)take((size_t)6 * HN_ * 4);
  float* EMBp  = (float*)take((size_t)BB * CC * 4);
  float* STATS = (float*)take((size_t)2048 * 4);
  u16*   Wb    = (u16*)take((size_t)1024 * 1024 * 2);
  u16*   Ab    = (u16*)take((size_t)MM * 768 * 2);
  float* Cf    = (float*)take((size_t)MM * 1024 * 4);
  float* X     = (float*)take((size_t)MM * HH * 4);
  float* U     = (float*)take((size_t)MM * HH * 4);
  float* P     = (float*)take((size_t)MM * HH * 4);

  const float invM = 1.0f / (float)MM;
  const dim3 tb(32, 8);

  auto run_s4 = [&](const float* par, const float* const* p) {
    // X (B,L,H) -> U (B,H,L)
    transpose_kernel<float, false><<<dim3(HH / 32, LL / 32, BB), tb, 0, stream>>>(X, U, LL, HH);
    scan_fwd<<<(BB * HH) / 8, 256, 0, stream>>>(U, par, p[5], P);     // y_fwd + u*D
    scan_bwd<<<(BB * HH) / 8, 256, 0, stream>>>(U, par, P);           // += y_bwd
    // gelu(P) (B,H,L) -> Ab bf16 (B,L,H)
    transpose_kernel<u16, true><<<dim3(LL / 32, HH / 32, BB), tb, 0, stream>>>(P, Ab, HH, LL);
    copy_bf16<<<(2 * HH * HH) / 256, 256, 0, stream>>>(p[6], Wb, 2 * HH * HH);  // W_out (2H,H) identity
    gemm_bf16<<<dim3((2 * HH) / 128, MM / 64), 256, 0, stream>>>(Ab, Wb, Cf, p[7], MM, 2 * HH, HH);
    glu_res_ln<<<MM, 256, 0, stream>>>(Cf, p[8], p[9], X);
  };

  // --- params ---
  discretize<<<HN_ / 256, 256, 0, stream>>>(s4a[0], s4a[1], s4a[2], s4a[3], s4a[4], PARa);
  discretize<<<HN_ / 256, 256, 0, stream>>>(s4b[0], s4b[1], s4b[2], s4b[3], s4b[4], PARb);

  // --- emb + conv3 + BN -> X ---
  emb_kernel<<<(BB * CC) / 256, 256, 0, stream>>>(demb, embW, embb, EMBp);
  pack_im2col<<<(MM * 768) / 256, 256, 0, stream>>>(x_t, EMBp, Ab);
  pack_w_mid<<<(HH * 768) / 256, 256, 0, stream>>>(midW, Wb);
  gemm_bf16<<<dim3(HH / 128, MM / 64), 256, 0, stream>>>(Ab, Wb, Cf, midb, MM, HH, 768);
  zero_kernel<<<8, 256, 0, stream>>>(STATS, 2048);
  bn_stats<<<dim3(HH / 256, MM / 128), 256, 0, stream>>>(Cf, STATS, HH);
  bn_apply_row<<<(MM * HH) / 256, 256, 0, stream>>>(Cf, STATS, midg, midbe, X, HH, invM, 0);

  // --- S4 layer a ---
  run_s4(PARa, s4a);

  // --- cond conv + BN, add into X ---
  transpose_kernel<u16, false><<<dim3(LL / 32, 128 / 32, BB), tb, 0, stream>>>(x_con, Ab, 128, LL);
  copy_bf16<<<(HH * 128) / 256, 256, 0, stream>>>(condW, Wb, HH * 128);
  gemm_bf16<<<dim3(HH / 128, MM / 64), 256, 0, stream>>>(Ab, Wb, Cf, condb, MM, HH, 128);
  zero_kernel<<<8, 256, 0, stream>>>(STATS, 2048);
  bn_stats<<<dim3(HH / 256, MM / 128), 256, 0, stream>>>(Cf, STATS, HH);
  bn_apply_row<<<(MM * HH) / 256, 256, 0, stream>>>(Cf, STATS, condg, condbe, X, HH, invM, 1);

  // --- S4 layer b ---
  run_s4(PARb, s4b);

  // --- GLU gate -> Ab bf16 (MM,256) ---
  glu_gate<<<(MM * CC) / 256, 256, 0, stream>>>(X, Ab);

  // --- res: 1x1 conv + BN, (x_t + bn)*sqrt(0.5) -> d_out[0:] (B,C,L) ---
  copy_bf16<<<(CC * CC) / 256, 256, 0, stream>>>(resW, Wb, CC * CC);
  gemm_bf16<<<dim3(CC / 128, MM / 64), 256, 0, stream>>>(Ab, Wb, Cf, resb, MM, CC, CC);
  zero_kernel<<<8, 256, 0, stream>>>(STATS, 2048);
  bn_stats<<<dim3(1, MM / 128), 256, 0, stream>>>(Cf, STATS, CC);
  bn_apply_tr<<<(BB * CC * LL) / 256, 256, 0, stream>>>(Cf, STATS, resg, resbe, x_t, dout,
                                                        CC, invM, 0.70710678118654752f);

  // --- skip: 1x1 conv + BN -> d_out[B*C*L:] (B,SKIP,L) ---
  copy_bf16<<<(CC * CC) / 256, 256, 0, stream>>>(skipW, Wb, CC * CC);
  gemm_bf16<<<dim3(CC / 128, MM / 64), 256, 0, stream>>>(Ab, Wb, Cf, skipb, MM, CC, CC);
  zero_kernel<<<8, 256, 0, stream>>>(STATS, 2048);
  bn_stats<<<dim3(1, MM / 128), 256, 0, stream>>>(Cf, STATS, CC);
  bn_apply_tr<<<(BB * CC * LL) / 256, 256, 0, stream>>>(Cf, STATS, skipg, skipbe, nullptr,
                                                        dout + (size_t)BB * CC * LL, CC, invM, 1.f);
}
